// LlamaAttentionKIVI_3169685865185
// MI455X (gfx1250) — matrix operand order repaired and checked by reading, not compiled, not run on hardware
//
#include <hip/hip_runtime.h>
#include <hip/hip_bf16.h>
#include <cstddef>

// ---------------------------------------------------------------------------
// LlamaAttention (KIVI residual KV cache) for gfx1250 / MI455X.
// - All matmuls: V_WMMA_F32_16X16X32_BF16 (fp32 accumulate).
// - GEMM tile staging: Tensor Data Mover (tensor_load_to_lds) with LDS
//   double-buffering; TDM pad_enable inserts bank-conflict padding.
// - fp32 -> bf16x2 packing via single v_perm_b32 at fragment build.
// - Attention: one-pass, 16x3712 score strip resident in LDS (~238KB of the
//   320KB WGP pool); 243MB attn_weights written exactly once.
// ---------------------------------------------------------------------------

typedef __attribute__((ext_vector_type(16))) __bf16 v16bf;
typedef __attribute__((ext_vector_type(2)))  __bf16 bf16x2;
typedef __attribute__((ext_vector_type(8)))  float  v8f;
typedef __attribute__((ext_vector_type(4)))  unsigned int u32x4;
typedef __attribute__((ext_vector_type(8)))  unsigned int u32x8;

#define SQ    512
#define HID   4096
#define NH    32
#define NKV   8
#define DHEAD 128
#define SC    3072
#define SRES  640          // 128 prev + 512 new
#define LQ    3712         // SC + SRES
#define SPAD  3720         // padded LDS row stride (floats)
#define QK_SCALE 0.08838834764831845f   // 1/sqrt(128)

union Frag { v16bf v; unsigned int u[8]; };

// pack two f32 -> bf16x2 dword: result = {hi16(f1), hi16(f0)} (one v_perm_b32)
static __device__ inline unsigned int pk2bf(float f1, float f0) {
  return __builtin_amdgcn_perm(__float_as_uint(f1), __float_as_uint(f0),
                               0x07060302u);
}

static __device__ inline __bf16 f2bf(float x) {   // RNE scalar convert
  unsigned int u = __float_as_uint(x);
  unsigned int r = (u + 0x7FFFu + ((u >> 16) & 1u)) >> 16;
  union { unsigned short u16; __bf16 b; } cv;
  cv.u16 = (unsigned short)r;
  return cv.b;
}

static __device__ inline v8f wmma_bf16(v16bf a, v16bf b, v8f c) {
  // (neg_a, A, neg_b, B, c_mod, C, reuse_a, reuse_b)
  return __builtin_amdgcn_wmma_f32_16x16x32_bf16(false, a, false, b,
                                                 (short)0, c, false, false);
}

// ---------------------------------------------------------------------------
// TDM: load a [rows x cols] fp32 tile (row stride = stride_elems) from global
// into LDS at lds_byte_addr, with 4-dword padding after every 32 dwords
// (matches the f32[rows][36] LDS layout). One instruction per tile; tracked
// with TENSORcnt.  D# layout per CDNA5 ISA 8.3/8.4.
// ---------------------------------------------------------------------------
static __device__ inline void tdm_load_tile_f32(unsigned int lds_byte_addr,
                                                const float* gptr,
                                                unsigned int rows,
                                                unsigned int cols,
                                                unsigned int stride_elems) {
  unsigned long long ga = (unsigned long long)(size_t)gptr;
  u32x4 g0;
  g0[0] = 1u;                                        // count=1, user descriptor
  g0[1] = lds_byte_addr;                             // lds_addr
  g0[2] = (unsigned int)(ga & 0xFFFFFFFFu);          // global_addr[31:0]
  g0[3] = (unsigned int)((ga >> 32) & 0x01FFFFFFu)   // global_addr[56:32]
          | (2u << 30);                              // type=2 (image)
  u32x8 g1;
  // data_size=4B(code2), pad_enable=1, pad_interval=32dw(code4), pad=4dw(code3)
  g1[0] = (2u << 16) | (1u << 20) | (4u << 22) | (3u << 25);
  g1[1] = (cols & 0xFFFFu) << 16;                    // tensor_dim0[15:0]
  g1[2] = ((cols >> 16) & 0xFFFFu) | ((rows & 0xFFFFu) << 16); // td0 hi | td1 lo
  g1[3] = ((rows >> 16) & 0xFFFFu) | ((cols & 0xFFFFu) << 16); // td1 hi | tile_dim0
  g1[4] = (rows & 0xFFFFu);                          // tile_dim1 | tile_dim2=0
  g1[5] = stride_elems;                              // tensor_dim0_stride[31:0]
  g1[6] = 0u;                                        // stride hi | dim1_stride lo
  g1[7] = 0u;
  asm volatile("tensor_load_to_lds %0, %1" :: "s"(g0), "s"(g1) : "memory");
}

// ---------------------------------------------------------------------------
// GEMM: C[M x N] = A[M x K] * W^T, W is [N x K] row-major (torch Linear).
// 64x64x32 block tiles, fp32 LDS tiles double-buffered via TDM, 8 waves x
// 2 WMMA tiles, perm-packed bf16 fragments.
// ---------------------------------------------------------------------------
#define BM 64
#define BN 64
#define BK 32
#define LPF 36   // padded f32 row stride (36*4B = 144B, 16B-multiple)

__global__ void __launch_bounds__(256)
gemm_xwt_bf16(const float* __restrict__ A, const float* __restrict__ W,
              float* __restrict__ C, int M, int N, int K) {
  __shared__ __align__(16) float As[2][BM][LPF];
  __shared__ __align__(16) float Bs[2][BN][LPF];
  const int tid  = threadIdx.x;
  const int lane = tid & 31;
  const int w    = tid >> 5;
  const int m0   = blockIdx.y * BM;
  const int n0   = blockIdx.x * BN;
  const int mt   = (w & 3) * 16;        // wave's M tile
  const int nt   = (w >> 2) * 32;       // wave's N tile pair base
  const int lrow = lane & 15;
  const int hi   = lane >> 4;           // half-wave select
  const int nk   = K / BK;
  v8f acc0 = {}; v8f acc1 = {};

  // prologue: TDM prefetch of k-block 0 into buffer 0 (wave 0 only)
  if (w == 0) {
    tdm_load_tile_f32((unsigned int)(size_t)&As[0][0][0],
                      A + (size_t)m0 * K, BM, BK, (unsigned int)K);
    tdm_load_tile_f32((unsigned int)(size_t)&Bs[0][0][0],
                      W + (size_t)n0 * K, BN, BK, (unsigned int)K);
  }

  for (int kb = 0; kb < nk; ++kb) {
    __syncthreads();                    // all waves done reading prev buffer
    const int cur = kb & 1;
    if (kb + 1 < nk) {
      if (w == 0) {                     // TDM prefetch next tile pair
        int k1 = (kb + 1) * BK;
        tdm_load_tile_f32((unsigned int)(size_t)&As[cur ^ 1][0][0],
                          A + (size_t)m0 * K + k1, BM, BK, (unsigned int)K);
        tdm_load_tile_f32((unsigned int)(size_t)&Bs[cur ^ 1][0][0],
                          W + (size_t)n0 * K + k1, BN, BK, (unsigned int)K);
      }
      __builtin_amdgcn_s_wait_tensorcnt((short)2);   // cur pair done, next in flight
    } else {
      __builtin_amdgcn_s_wait_tensorcnt((short)0);
    }
    __syncthreads();                    // cur buffer visible to all waves

    Frag a, b0, b1;
#pragma unroll
    for (int t = 0; t < 8; ++t) {
      // A frag, this code's contiguous k order (B uses the same): lanes0-15 K=0..15, lanes16-31 K=16..31
      int ka = 2 * t + hi * 16;
      float2 fa = *(const float2*)&As[cur][mt + lrow][ka];
      a.u[t] = pk2bf(fa.y, fa.x);
      // B frag, same contiguous k order as A: lanes0-15 K=0..15, lanes16-31 K=16..31, N=lane%16
      int kb2 = 2 * t + hi * 16;
      float2 f0 = *(const float2*)&Bs[cur][nt + lrow][kb2];
      b0.u[t] = pk2bf(f0.y, f0.x);
      float2 f1 = *(const float2*)&Bs[cur][nt + 16 + lrow][kb2];
      b1.u[t] = pk2bf(f1.y, f1.x);
    }
    acc0 = wmma_bf16(a.v, b0.v, acc0);
    acc1 = wmma_bf16(a.v, b1.v, acc1);
  }
#pragma unroll
  for (int r = 0; r < 8; ++r) {
    int row = m0 + mt + r + hi * 8;     // C/D: VGPR r -> M=r (+8 for hi half)
    C[(size_t)row * N + n0 + nt + lrow]      = acc0[r];
    C[(size_t)row * N + n0 + nt + 16 + lrow] = acc1[r];
  }
}

// ---------------------------------------------------------------------------
// RoPE on Q (folds in 1/sqrt(D) scale).  q: [512 x 4096] = [s][h*128+d]
// ---------------------------------------------------------------------------
__global__ void __launch_bounds__(256)
rope_scale_q(const float* __restrict__ qraw, const float* __restrict__ cosb,
             const float* __restrict__ sinb, float* __restrict__ qout) {
  int idx = blockIdx.x * 256 + threadIdx.x;   // 512*4096 total
  int s = idx >> 12;
  int d = idx & 127;
  float x   = qraw[idx];
  float rot = (d < 64) ? -qraw[idx + 64] : qraw[idx - 64];
  float val = x * cosb[s * DHEAD + d] + rot * sinb[s * DHEAD + d];
  qout[idx] = val * QK_SCALE;
}

// ---------------------------------------------------------------------------
// Build residual caches: k_res/v_res [8 x 640 x 128]; j<128 from prev,
// j>=128 from projected K (with RoPE) / V.
// ---------------------------------------------------------------------------
__global__ void __launch_bounds__(256)
build_residual(const float* __restrict__ kraw, const float* __restrict__ vraw,
               const float* __restrict__ kprev, const float* __restrict__ vprev,
               const float* __restrict__ cosb, const float* __restrict__ sinb,
               float* __restrict__ kres, float* __restrict__ vres) {
  int idx = blockIdx.x * 256 + threadIdx.x;   // 8*640*128 total
  int d   = idx & 127;
  int t   = idx >> 7;
  int j   = t % SRES;
  int kvh = t / SRES;
  float kv, vv;
  if (j < 128) {
    int src = (kvh * 128 + j) * DHEAD + d;
    kv = kprev[src];
    vv = vprev[src];
  } else {
    int s = j - 128;
    int src = s * (NKV * DHEAD) + kvh * DHEAD + d;
    float x   = kraw[src];
    float rot = (d < 64) ? -kraw[src + 64] : kraw[src - 64];
    kv = x * cosb[s * DHEAD + d] + rot * sinb[s * DHEAD + d];
    vv = vraw[src];
  }
  kres[idx] = kv;
  vres[idx] = vv;
}

// ---------------------------------------------------------------------------
// Attention: one workgroup per (head, 16-row query tile).
// Dynamic LDS: S[16 x SPAD] f32 scores + red/red2[16x16] + Qb[16x128] bf16.
// ---------------------------------------------------------------------------
__global__ void __launch_bounds__(256)
attn_kernel(const float* __restrict__ qs, const float* __restrict__ kcache,
            const float* __restrict__ vcache, const float* __restrict__ kres,
            const float* __restrict__ vres, const float* __restrict__ mask,
            float* __restrict__ attn_w, float* __restrict__ ctx) {
  extern __shared__ char smem[];
  float*  S    = (float*)smem;                               // 16*SPAD
  float*  red  = (float*)(smem + (size_t)16 * SPAD * 4);     // 16*16
  float*  red2 = red + 256;                                  // 16*16
  __bf16* Qb   = (__bf16*)(red2 + 256);                      // 16*128

  const int h    = blockIdx.x;          // 0..31
  const int qt   = blockIdx.y;          // 0..31
  const int kvh  = h >> 2;              // GROUPS = 4
  const int s0   = qt * 16;
  const int tid  = threadIdx.x;
  const int lane = tid & 31;
  const int w    = tid >> 5;
  const int lrow = lane & 15;
  const int hi   = lane >> 4;

  // Stage scaled+roped Q tile as bf16 (RNE, done once per element)
  for (int i = tid; i < 16 * DHEAD; i += 256) {
    int r = i >> 7, d = i & 127;
    Qb[i] = f2bf(qs[(size_t)(s0 + r) * HID + h * DHEAD + d]);
  }
  __syncthreads();

  // ---- scores: S[16 x 3712] = Q Kt + mask (Q pre-scaled) ----
  for (int ct = w; ct < LQ / 16; ct += 8) {
    int j0 = ct * 16;
    int j  = j0 + lrow;
    const float* krow = (j < SC)
        ? (kcache + ((size_t)kvh * SC + j) * DHEAD)
        : (kres   + ((size_t)kvh * SRES + (j - SC)) * DHEAD);
    v8f acc = {};
#pragma unroll
    for (int k0 = 0; k0 < DHEAD; k0 += 32) {
      Frag a, b;
#pragma unroll
      for (int t = 0; t < 8; ++t) {
        int ka = k0 + 2 * t + hi * 16;
        bf16x2 pa = *(const bf16x2*)&Qb[lrow * DHEAD + ka];
        a.v[2 * t] = pa.x; a.v[2 * t + 1] = pa.y;
      }
      const float4* kv4 = (const float4*)(krow + k0 + hi * 16);
#pragma unroll
      for (int t = 0; t < 4; ++t) {
        float4 f = kv4[t];
        b.u[2 * t]     = pk2bf(f.y, f.x);
        b.u[2 * t + 1] = pk2bf(f.w, f.z);
      }
      acc = wmma_bf16(a.v, b.v, acc);
    }
#pragma unroll
    for (int r = 0; r < 8; ++r) {
      int row = r + hi * 8;
      int col = j0 + lrow;
      S[row * SPAD + col] = acc[r] + mask[(size_t)(s0 + row) * LQ + col];
    }
  }
  __syncthreads();

  // ---- softmax over each of the 16 rows (16 threads per row) ----
  {
    int row = tid >> 4;
    int sub = tid & 15;
    float m = -3.4e38f;
    for (int j = sub; j < LQ; j += 16) m = fmaxf(m, S[row * SPAD + j]);
    red[row * 16 + sub] = m;
    __syncthreads();
    float rm = red[row * 16];
#pragma unroll
    for (int i = 1; i < 16; ++i) rm = fmaxf(rm, red[row * 16 + i]);
    float lsum = 0.f;
    for (int j = sub; j < LQ; j += 16) {
      float e = __expf(S[row * SPAD + j] - rm);
      S[row * SPAD + j] = e;
      lsum += e;
    }
    red2[row * 16 + sub] = lsum;
    __syncthreads();
    float tot = 0.f;
#pragma unroll
    for (int i = 0; i < 16; ++i) tot += red2[row * 16 + i];
    float inv = 1.f / tot;
    float* orow = attn_w + ((size_t)h * SQ + s0 + row) * LQ;
    for (int j = sub; j < LQ; j += 16) {
      float p = S[row * SPAD + j] * inv;
      S[row * SPAD + j] = p;
      orow[j] = p;                       // single coalesced HBM write
    }
  }
  __syncthreads();

  // ---- context: ctx[16 x 128] = P V ; wave w owns 16-col slice ----
  {
    int d0 = w * 16;
    v8f acc = {};
    for (int j0 = 0; j0 < LQ; j0 += 32) {
      Frag a, b;
#pragma unroll
      for (int t = 0; t < 8; ++t) {
        int jj = j0 + 2 * t + hi * 16;
        float2 f = *(const float2*)&S[lrow * SPAD + jj];
        a.u[t] = pk2bf(f.y, f.x);
      }
#pragma unroll
      for (int t = 0; t < 8; ++t) {
        int j = j0 + 2 * t + hi * 16;
        const float* vr0 = (j < SC)
            ? (vcache + ((size_t)kvh * SC + j) * DHEAD)
            : (vres   + ((size_t)kvh * SRES + (j - SC)) * DHEAD);
        const float* vr1 = (j + 1 < SC)
            ? (vcache + ((size_t)kvh * SC + j + 1) * DHEAD)
            : (vres   + ((size_t)kvh * SRES + (j + 1 - SC)) * DHEAD);
        b.u[t] = pk2bf(vr1[d0 + lrow], vr0[d0 + lrow]);
      }
      acc = wmma_bf16(a.v, b.v, acc);
    }
#pragma unroll
    for (int r = 0; r < 8; ++r) {
      int row = s0 + r + hi * 8;
      ctx[(size_t)row * HID + h * DHEAD + d0 + lrow] = acc[r];
    }
  }
}

// ---------------------------------------------------------------------------
extern "C" void kernel_launch(void* const* d_in, const int* in_sizes, int n_in,
                              void* d_out, int out_size, void* d_ws, size_t ws_size,
                              hipStream_t stream) {
  const float* hidden = (const float*)d_in[0];
  const float* cosb   = (const float*)d_in[1];
  const float* sinb   = (const float*)d_in[2];
  const float* kcache = (const float*)d_in[3];
  const float* vcache = (const float*)d_in[4];
  const float* kprev  = (const float*)d_in[5];
  const float* vprev  = (const float*)d_in[6];
  const float* maskp  = (const float*)d_in[7];
  const float* Wq     = (const float*)d_in[8];
  const float* Wk     = (const float*)d_in[9];
  const float* Wv     = (const float*)d_in[10];
  const float* Wo     = (const float*)d_in[11];
  float* out = (float*)d_out;
  float* ws  = (float*)d_ws;

  // workspace carve-up (floats); total ~33 MB
  float* q_raw = ws;
  float* k_raw = q_raw + (size_t)SQ * HID;            // 512*4096
  float* v_raw = k_raw + (size_t)SQ * NKV * DHEAD;    // 512*1024
  float* q_rs  = v_raw + (size_t)SQ * NKV * DHEAD;
  float* k_res = q_rs  + (size_t)SQ * HID;
  float* v_res = k_res + (size_t)NKV * SRES * DHEAD;  // 8*640*128
  float* ctx   = v_res + (size_t)NKV * SRES * DHEAD;
  float* attn_w = out + (size_t)SQ * HID;             // d_out part 2

  dim3 blk(256);
  // QKV projections
  gemm_xwt_bf16<<<dim3(HID / BN, SQ / BM), blk, 0, stream>>>(hidden, Wq, q_raw, SQ, HID, HID);
  gemm_xwt_bf16<<<dim3((NKV * DHEAD) / BN, SQ / BM), blk, 0, stream>>>(hidden, Wk, k_raw, SQ, NKV * DHEAD, HID);
  gemm_xwt_bf16<<<dim3((NKV * DHEAD) / BN, SQ / BM), blk, 0, stream>>>(hidden, Wv, v_raw, SQ, NKV * DHEAD, HID);
  // RoPE + residual cache assembly
  rope_scale_q<<<dim3((SQ * HID) / 256), blk, 0, stream>>>(q_raw, cosb, sinb, q_rs);
  build_residual<<<dim3((NKV * SRES * DHEAD) / 256), blk, 0, stream>>>(
      k_raw, v_raw, kprev, vprev, cosb, sinb, k_res, v_res);
  // Attention (big-LDS single pass)
  size_t smem = (size_t)16 * SPAD * 4 + 2 * 256 * 4 + (size_t)16 * DHEAD * 2;
  attn_kernel<<<dim3(NH, SQ / 16), blk, smem, stream>>>(
      q_rs, kcache, vcache, k_res, v_res, maskp, attn_w, ctx);
  // Output projection
  gemm_xwt_bf16<<<dim3(HID / BN, SQ / BM), blk, 0, stream>>>(ctx, Wo, out, SQ, HID, HID);
}
